// MambaBlock_35613868818667
// MI455X (gfx1250) — matrix operand order repaired and checked by reading, not compiled, not run on hardware
//
#include <hip/hip_runtime.h>
#include <hip/hip_bf16.h>
#include <math.h>

// ---------------- model constants ----------------
#define CDIM   512
#define DIN    1024
#define NST    16
#define DTRK   32
#define BB     2
#define TT     1024
#define NTOT   (BB*TT)        // 2048
#define SCALE_ 0.1f

typedef __attribute__((ext_vector_type(16))) _Float16 v16h;
typedef __attribute__((ext_vector_type(8)))  _Float16 v8h;
typedef __attribute__((ext_vector_type(8)))  float    v8f;

// LDS row stride in halves: 40 halves = 80 B = 20 banks -> the 16 row-parallel
// ds_load_b128 fragment reads cover all 64 LDS banks exactly once (no conflicts).
#define ASTR 40

// ---------------- device helpers ----------------
__device__ __forceinline__ float gelu_f(float x) {
    return 0.5f * x * (1.0f + erff(x * 0.70710678118654752f));
}
__device__ __forceinline__ float softplus_f(float x) {
    return (x > 20.0f) ? x : log1pf(__expf(x));
}
__device__ __forceinline__ float silu_f(float x) {
    return x / (1.0f + __expf(-x));
}
__device__ __forceinline__ v8h cvt8(float4 a, float4 b) {
    v8h h;
    h[0] = (_Float16)a.x; h[1] = (_Float16)a.y; h[2] = (_Float16)a.z; h[3] = (_Float16)a.w;
    h[4] = (_Float16)b.x; h[5] = (_Float16)b.y; h[6] = (_Float16)b.z; h[7] = (_Float16)b.w;
    return h;
}

// =================================================================
// Generic tiled WMMA GEMM:   Y[M][N] = act(A[M][K] * X[K][N] + bias)
//   mode 0: X k-major, rows k<ksplit from X0 else X1 (concat along K).
//           ksplit is a multiple of 32, so a whole 32-wide K-tile always
//           lives on one side -> base pointer selected once per tile.
//   mode 1: gram  Y = (A . A^T) scaled by aux[m]*aux[n]   (A is [M][K], ldA)
//   mode 2: like mode 0 but epilogue Y = (1+S)*aux[m*ldn+n] - S*acc
// Tile: 128(M) x 64(N), K-step 32, 256 threads = 8 wave32 waves (4x2),
// each wave does a 2x2 grid of 16x16x32 f16 WMMAs.
// Double-buffered LDS: iteration k+1's global loads are issued before
// iteration k's WMMAs; one workgroup barrier per K-step.
// All M multiples of 16, N multiples of 64, K multiples of 32.
// =================================================================
__global__ __launch_bounds__(256)
__attribute__((amdgpu_waves_per_eu(2, 8)))
void wmma_gemm(const float* __restrict__ A, int ldA,
               const float* __restrict__ X0, const float* __restrict__ X1,
               int ksplit, int ldn,
               const float* __restrict__ bias,
               const float* __restrict__ aux,
               float* __restrict__ Y, int ldy,
               int M, int N, int K, int act, int mode)
{
    __shared__ __align__(16) _Float16 As[2][128 * ASTR];
    __shared__ __align__(16) _Float16 Bs[2][64 * ASTR];

    const int tid  = threadIdx.x;
    const int row0 = blockIdx.y * 128;
    const int col0 = blockIdx.x * 64;
    const int wave = tid >> 5, lane = tid & 31;
    const int wm   = wave & 3, wn = wave >> 2;        // wm 0..3, wn 0..1
    const int lr   = lane & 15, lh = lane >> 4;

    // staging decompositions (fixed per thread)
    const int sa_m  = tid >> 1;            // 0..127
    const int sa_kk = (tid & 1) * 16;      // 0 / 16
    const int sb_n  = tid & 63;            // mode0: n lane
    const int sb_g  = tid >> 6;            // mode0: k-group 0..3
    const int sg_n  = tid >> 2;            // mode1: 0..63
    const int sg_kk = (tid & 3) * 8;       // mode1: 0/8/16/24

    const bool aFull = (row0 + 128 <= M);  // uniform: all GEMMs except M=64 proj

    v8f acc[2][2] = {};

    // prefetch registers (fp32 kept live across the WMMA block so the
    // loadcnt waits land after the matrix ops)
    float4 ra0{}, ra1{}, ra2{}, ra3{};
    float4 rb0{}, rb1{};
    float  rf[8];

    auto load_tile = [&](int k0) {
        if (aFull) {                        // uniform branch, no lane merges
            const float4* src = (const float4*)(A + (size_t)(row0 + sa_m) * ldA + k0 + sa_kk);
            ra0 = src[0]; ra1 = src[1]; ra2 = src[2]; ra3 = src[3];
        } else {
            ra0 = ra1 = ra2 = ra3 = float4{0, 0, 0, 0};
            if ((row0 + sa_m) < M) {
                const float4* src = (const float4*)(A + (size_t)(row0 + sa_m) * ldA + k0 + sa_kk);
                ra0 = src[0]; ra1 = src[1]; ra2 = src[2]; ra3 = src[3];
            }
        }
        if (mode == 1) {
            const float4* src = (const float4*)(A + (size_t)(col0 + sg_n) * ldA + k0 + sg_kk);
            rb0 = src[0]; rb1 = src[1];
        } else {
            // whole 32-wide tile is on one side of the K split (uniform select)
            const float* base = (k0 < ksplit)
                ? (X0 + (size_t)k0 * ldn)
                : (X1 + (size_t)(k0 - ksplit) * ldn);
            const float* src = base + (size_t)(sb_g * 8) * ldn + col0 + sb_n;
            #pragma unroll
            for (int i = 0; i < 8; ++i)
                rf[i] = src[(size_t)i * ldn];
        }
    };
    auto store_tile = [&](int buf) {
        *(v8h*)&As[buf][sa_m * ASTR + sa_kk]     = cvt8(ra0, ra1);
        *(v8h*)&As[buf][sa_m * ASTR + sa_kk + 8] = cvt8(ra2, ra3);
        if (mode == 1) {
            *(v8h*)&Bs[buf][sg_n * ASTR + sg_kk] = cvt8(rb0, rb1);
        } else {
            v8h h;
            #pragma unroll
            for (int i = 0; i < 8; ++i) h[i] = (_Float16)rf[i];
            *(v8h*)&Bs[buf][sb_n * ASTR + sb_g * 8] = h;
        }
    };

    const int ktiles = K >> 5;
    load_tile(0);
    store_tile(0);
    __syncthreads();

    for (int kt = 0; kt < ktiles; ++kt) {
        const int cur = kt & 1;
        const bool more = (kt + 1 < ktiles);
        if (more) load_tile((kt + 1) << 5);   // overlap with WMMAs below

        // B fragments first, then stream A fragments one at a time to
        // minimize peak fragment liveness (24 VGPRs instead of 32).
        v16h bfr[2];
        #pragma unroll
        for (int tn = 0; tn < 2; ++tn) {
            const _Float16* pb = &Bs[cur][(wn * 32 + tn * 16 + lr) * ASTR];
            v8h lo = *(const v8h*)(pb + lh * 8);
            v8h hi = *(const v8h*)(pb + 16 + lh * 8);
            #pragma unroll
            for (int j = 0; j < 8; ++j) { bfr[tn][j] = lo[j]; bfr[tn][j + 8] = hi[j]; }
        }
        #pragma unroll
        for (int tm = 0; tm < 2; ++tm) {
            const _Float16* pa = &As[cur][(wm * 32 + tm * 16 + lr) * ASTR];
            v8h lo = *(const v8h*)(pa + lh * 8);
            v8h hi = *(const v8h*)(pa + 16 + lh * 8);
            v16h a;
            #pragma unroll
            for (int j = 0; j < 8; ++j) { a[j] = lo[j]; a[j + 8] = hi[j]; }
            acc[tm][0] = __builtin_amdgcn_wmma_f32_16x16x32_f16(
                false, a, false, bfr[0], (short)0, acc[tm][0], false, false);
            acc[tm][1] = __builtin_amdgcn_wmma_f32_16x16x32_f16(
                false, a, false, bfr[1], (short)0, acc[tm][1], false, false);
        }

        if (more) store_tile(cur ^ 1);
        __syncthreads();
    }

    // ---- epilogue (guard at 16-row tile granularity) ----
    #pragma unroll
    for (int tm = 0; tm < 2; ++tm) {
        const int mbase = row0 + wm * 32 + tm * 16;
        if (mbase >= M) continue;
        #pragma unroll
        for (int tn = 0; tn < 2; ++tn) {
            const int n = col0 + wn * 32 + tn * 16 + lr;
            #pragma unroll
            for (int j = 0; j < 8; ++j) {
                const int m = mbase + lh * 8 + j;
                float v = acc[tm][tn][j];
                if (mode == 1) {
                    v = v * aux[m] * aux[n];
                } else {
                    if (bias) v += bias[m];
                    if (act == 1)      v = gelu_f(v);
                    else if (act == 2) v = fmaxf(v, 0.0f);
                    else if (act == 3) v = softplus_f(v);
                    if (mode == 2)
                        v = (1.0f + SCALE_) * aux[(size_t)m * ldn + n] - SCALE_ * v;
                }
                Y[(size_t)m * ldy + n] = v;
            }
        }
    }
}

// =================================================================
// Elementwise / reduction kernels
// =================================================================

// repack w_ds (O,C,3) -> [O][k*C + c]
__global__ void repack_wds(const float* __restrict__ w, float* __restrict__ out) {
    int gid = blockIdx.x * blockDim.x + threadIdx.x;
    if (gid >= CDIM * CDIM * 3) return;
    int o = gid / (CDIM * 3);
    int r = gid - o * (CDIM * 3);
    int c = r / 3, k = r - c * 3;
    out[(size_t)o * (3 * CDIM) + k * CDIM + c] = w[gid];
}

// builds pooled[c][n] and im2col xds[(k*C+c)][n]
__global__ void prep_kernel(const float* __restrict__ time,
                            float* __restrict__ pool, float* __restrict__ xds) {
    int gid = blockIdx.x * blockDim.x + threadIdx.x;
    if (gid >= CDIM * NTOT) return;
    int c = gid >> 11, n = gid & (NTOT - 1);
    int b = n >> 10, t = n & (TT - 1);
    const float* tp = time + ((size_t)b * CDIM + c) * TT;
    float mid = tp[t];
    float m = mid;
    float lft = 0.0f, rgt = 0.0f;
    if (t > 0)      { lft = tp[t - 1]; m = fmaxf(m, lft); }
    if (t < TT - 1) { rgt = tp[t + 1]; m = fmaxf(m, rgt); }
    pool[(size_t)c * NTOT + n] = m;
    xds[((size_t)0 * CDIM + c) * NTOT + n] = lft;
    xds[((size_t)1 * CDIM + c) * NTOT + n] = mid;
    xds[((size_t)2 * CDIM + c) * NTOT + n] = rgt;
}

// layernorm over channel dim per position n
__global__ void ln_channels(const float* __restrict__ x, const float* __restrict__ g,
                            const float* __restrict__ b, float* __restrict__ y) {
    int n = blockIdx.x;
    __shared__ float s1[256], s2[256];
    float a = 0.f, q = 0.f;
    for (int c = threadIdx.x; c < CDIM; c += 256) {
        float v = x[(size_t)c * NTOT + n]; a += v; q += v * v;
    }
    s1[threadIdx.x] = a; s2[threadIdx.x] = q; __syncthreads();
    for (int s = 128; s > 0; s >>= 1) {
        if (threadIdx.x < s) { s1[threadIdx.x] += s1[threadIdx.x + s]; s2[threadIdx.x] += s2[threadIdx.x + s]; }
        __syncthreads();
    }
    float mu  = s1[0] / CDIM;
    float var = s2[0] / CDIM - mu * mu;
    float inv = rsqrtf(var + 1e-5f);
    for (int c = threadIdx.x; c < CDIM; c += 256) {
        float v = x[(size_t)c * NTOT + n];
        y[(size_t)c * NTOT + n] = (v - mu) * inv * g[c] + b[c];
    }
}

// causal depthwise conv (K=4) + SiLU; rev=1 processes reversed time order
__global__ void conv_silu(const float* __restrict__ xin, const float* __restrict__ w,
                          const float* __restrict__ bb, float* __restrict__ xc, int rev) {
    int gid = blockIdx.x * blockDim.x + threadIdx.x;
    if (gid >= DIN * NTOT) return;
    int d = gid >> 11, n = gid & (NTOT - 1);
    int b = n >> 10, t = n & (TT - 1);
    const float* xr = xin + (size_t)d * NTOT + b * TT;
    float acc = bb[d];
    #pragma unroll
    for (int k = 0; k < 4; ++k) {
        int tt = t + k - 3;
        if (tt >= 0) {
            int src = rev ? (TT - 1 - tt) : tt;
            acc += w[d * 4 + k] * xr[src];
        }
    }
    xc[(size_t)d * NTOT + n] = silu_f(acc);
}

// selective scan: one thread per (b, d); 16 states in registers
__global__ void scan_kernel(const float* __restrict__ xc, const float* __restrict__ delta,
                            const float* __restrict__ proj, const float* __restrict__ A_log,
                            const float* __restrict__ Dp, float* __restrict__ y) {
    int gid = blockIdx.x * blockDim.x + threadIdx.x;
    if (gid >= BB * DIN) return;
    int b = gid >> 10, d = gid & (DIN - 1);
    float An[NST], h[NST];
    #pragma unroll
    for (int s = 0; s < NST; ++s) { An[s] = -__expf(A_log[d * NST + s]); h[s] = 0.0f; }
    float Dd = Dp[d];
    const size_t drow = (size_t)d * NTOT;
    for (int t = 0; t < TT; ++t) {
        int n = b * TT + t;
        float dl = delta[drow + n];
        float u  = xc[drow + n];
        float du = dl * u;
        float yt = 0.0f;
        #pragma unroll
        for (int s = 0; s < NST; ++s) {
            float dA = __expf(dl * An[s]);
            h[s] = dA * h[s] + du * proj[(size_t)(DTRK + s) * NTOT + n];
            yt  += h[s] * proj[(size_t)(DTRK + NST + s) * NTOT + n];
        }
        y[drow + n] = yt + Dd * u;
    }
}

// ycomb = (yf + yb(time-reversed)) * silu(z)
__global__ void combine_kernel(const float* __restrict__ yf, const float* __restrict__ yb,
                               const float* __restrict__ z, float* __restrict__ out) {
    int gid = blockIdx.x * blockDim.x + threadIdx.x;
    if (gid >= DIN * NTOT) return;
    int d = gid >> 11, n = gid & (NTOT - 1);
    int b = n >> 10, t = n & (TT - 1);
    float s = silu_f(z[(size_t)d * NTOT + n]);
    float vb = yb[(size_t)d * NTOT + b * TT + (TT - 1 - t)];
    out[(size_t)d * NTOT + n] = (yf[(size_t)d * NTOT + n] + vb) * s;
}

// low = (t2 + mout) * mask   (in place into t2)
__global__ void low_kernel(float* __restrict__ t2, const float* __restrict__ mout,
                           const unsigned char* __restrict__ mask) {
    int gid = blockIdx.x * blockDim.x + threadIdx.x;
    if (gid >= CDIM * NTOT) return;
    int n = gid & (NTOT - 1);
    float mf = mask[n] ? 1.0f : 0.0f;
    t2[gid] = (t2[gid] + mout[gid]) * mf;
}

// per-row (b,c) inverse L2 norm over t of out0
__global__ void rownorm_kernel(const float* __restrict__ x, float* __restrict__ rninv) {
    int r = blockIdx.x;                 // r = b*512 + c
    int b = r >> 9, c = r & (CDIM - 1);
    const float* src = x + (size_t)c * NTOT + b * TT;
    __shared__ float s1[256];
    float q = 0.f;
    for (int t = threadIdx.x; t < TT; t += 256) { float v = src[t]; q += v * v; }
    s1[threadIdx.x] = q; __syncthreads();
    for (int s = 128; s > 0; s >>= 1) {
        if (threadIdx.x < s) s1[threadIdx.x] += s1[threadIdx.x + s];
        __syncthreads();
    }
    if (threadIdx.x == 0) rninv[r] = 1.0f / fmaxf(sqrtf(s1[0]), 1e-12f);
}

// row softmax over 512 columns, in place
__global__ void softmax_rows(float* __restrict__ G) {
    int r = blockIdx.x;
    float* row = G + (size_t)r * CDIM;
    __shared__ float s1[256];
    float mx = -1e30f;
    for (int k = threadIdx.x; k < CDIM; k += 256) mx = fmaxf(mx, row[k]);
    s1[threadIdx.x] = mx; __syncthreads();
    for (int s = 128; s > 0; s >>= 1) {
        if (threadIdx.x < s) s1[threadIdx.x] = fmaxf(s1[threadIdx.x], s1[threadIdx.x + s]);
        __syncthreads();
    }
    mx = s1[0]; __syncthreads();
    float sum = 0.f;
    for (int k = threadIdx.x; k < CDIM; k += 256) { float e = __expf(row[k] - mx); row[k] = e; sum += e; }
    s1[threadIdx.x] = sum; __syncthreads();
    for (int s = 128; s > 0; s >>= 1) {
        if (threadIdx.x < s) s1[threadIdx.x] += s1[threadIdx.x + s];
        __syncthreads();
    }
    float inv = 1.0f / s1[0];
    for (int k = threadIdx.x; k < CDIM; k += 256) row[k] *= inv;
}

// final per-(b,c) norm over t with cn_g[t], cn_b[t]; writes d_out (B,C,T)
__global__ void final_norm(const float* __restrict__ x, const float* __restrict__ g,
                           const float* __restrict__ b2, float* __restrict__ out) {
    int r = blockIdx.x;                 // b*512 + c
    int b = r >> 9, c = r & (CDIM - 1);
    const float* src = x + (size_t)c * NTOT + b * TT;
    __shared__ float s1[256], s2[256];
    float a = 0.f, q = 0.f;
    for (int t = threadIdx.x; t < TT; t += 256) { float v = src[t]; a += v; q += v * v; }
    s1[threadIdx.x] = a; s2[threadIdx.x] = q; __syncthreads();
    for (int s = 128; s > 0; s >>= 1) {
        if (threadIdx.x < s) { s1[threadIdx.x] += s1[threadIdx.x + s]; s2[threadIdx.x] += s2[threadIdx.x + s]; }
        __syncthreads();
    }
    float mu  = s1[0] / TT;
    float var = s2[0] / TT - mu * mu;
    float inv = rsqrtf(var + 1e-6f);
    float* dst = out + ((size_t)b * CDIM + c) * TT;
    for (int t = threadIdx.x; t < TT; t += 256)
        dst[t] = (src[t] - mu) * inv * g[t] + b2[t];
}

__global__ void mask_out_kernel(const unsigned char* __restrict__ mask, float* __restrict__ out) {
    int i = blockIdx.x * blockDim.x + threadIdx.x;
    if (i < BB * TT) out[i] = mask[i] ? 1.0f : 0.0f;
}

// =================================================================
// Host orchestration
// =================================================================
static inline dim3 gemm_grid(int M, int N) { return dim3(N / 64, (M + 127) / 128); }

extern "C" void kernel_launch(void* const* d_in, const int* in_sizes, int n_in,
                              void* d_out, int out_size, void* d_ws, size_t ws_size,
                              hipStream_t stream) {
    const float* I_time   = (const float*)d_in[0];
    const float* I_wds    = (const float*)d_in[1];
    const float* I_bds    = (const float*)d_in[2];
    const float* I_wpc    = (const float*)d_in[3];
    const float* I_bpc    = (const float*)d_in[4];
    const float* I_lng    = (const float*)d_in[5];
    const float* I_lnb    = (const float*)d_in[6];
    const float* I_win    = (const float*)d_in[7];
    const float* I_wconv[2]  = {(const float*)d_in[8],  (const float*)d_in[15]};
    const float* I_bconv[2]  = {(const float*)d_in[9],  (const float*)d_in[16]};
    const float* I_wxproj[2] = {(const float*)d_in[10], (const float*)d_in[17]};
    const float* I_wdt[2]    = {(const float*)d_in[11], (const float*)d_in[18]};
    const float* I_bdt[2]    = {(const float*)d_in[12], (const float*)d_in[19]};
    const float* I_Alog[2]   = {(const float*)d_in[13], (const float*)d_in[20]};
    const float* I_D[2]      = {(const float*)d_in[14], (const float*)d_in[21]};
    const float* I_wout  = (const float*)d_in[22];
    const float* I_bout  = (const float*)d_in[23];
    const float* I_wc1   = (const float*)d_in[24];
    const float* I_bc1   = (const float*)d_in[25];
    const float* I_wc2   = (const float*)d_in[26];
    const float* I_bc2   = (const float*)d_in[27];
    const float* I_wc3   = (const float*)d_in[28];
    const float* I_bc3   = (const float*)d_in[29];
    const float* I_cng   = (const float*)d_in[30];
    const float* I_cnb   = (const float*)d_in[31];
    const unsigned char* I_mask = (const unsigned char*)d_in[32];

    float* ws = (float*)d_ws;
    // ---- workspace layout (floats), with reuse ----
    size_t off = 0;
    float* WDSP  = ws + off; off += (size_t)CDIM * 3 * CDIM;      //  786,432
    float* XDS   = ws + off; off += (size_t)3 * CDIM * NTOT;      // 3,145,728 (reused as XC)
    float* POOL  = ws + off; off += (size_t)CDIM * NTOT;          // 1,048,576 (reused as XN)
    float* HIGH  = ws + off; off += (size_t)CDIM * NTOT;
    float* T2    = ws + off; off += (size_t)CDIM * NTOT;          // becomes LOW in place
    float* XZ    = ws + off; off += (size_t)2 * DIN * NTOT;       // xin|z, later HIGH2|LOW2
    float* DELTA = ws + off; off += (size_t)DIN * NTOT;           // later YCOMB
    float* YF    = ws + off; off += (size_t)DIN * NTOT;           // later G/sim
    float* YB    = ws + off; off += (size_t)DIN * NTOT;           // later OUT1
    float* MOUT  = ws + off; off += (size_t)CDIM * NTOT;          // later OUT0
    float* PROJ  = ws + off; off += (size_t)64 * NTOT;
    float* RNINV = ws + off; off += 1024;
    (void)ws_size; (void)n_in; (void)in_sizes; (void)out_size;

    float* XC    = XDS;                 // reuse after t2 GEMM
    float* XN    = POOL;                // reuse after high GEMM
    float* XIN   = XZ;
    float* Z     = XZ + (size_t)DIN * NTOT;
    float* YCOMB = DELTA;
    float* HIGH2 = XZ;
    float* LOW2  = XZ + (size_t)DIN * NTOT;
    float* OUT0  = MOUT;
    float* G     = YF;
    float* OUT1  = YB;
    float* LOW   = T2;
    float* outf  = (float*)d_out;

    const int TPB = 256;
    // 1) repack w_ds, build pooled + im2col
    repack_wds<<<(CDIM * CDIM * 3 + TPB - 1) / TPB, TPB, 0, stream>>>(I_wds, WDSP);
    prep_kernel<<<(CDIM * NTOT + TPB - 1) / TPB, TPB, 0, stream>>>(I_time, POOL, XDS);

    // 2) high = w_pc * pooled + b_pc
    wmma_gemm<<<gemm_grid(CDIM, NTOT), TPB, 0, stream>>>(
        I_wpc, CDIM, POOL, POOL, 1 << 30, NTOT, I_bpc, nullptr, HIGH, NTOT,
        CDIM, NTOT, CDIM, 0, 0);
    // 3) t2 = gelu(Wds_packed * xds + b_ds)
    wmma_gemm<<<gemm_grid(CDIM, NTOT), TPB, 0, stream>>>(
        WDSP, 3 * CDIM, XDS, XDS, 1 << 30, NTOT, I_bds, nullptr, T2, NTOT,
        CDIM, NTOT, 3 * CDIM, 1, 0);
    // 4) layernorm over channels
    ln_channels<<<NTOT, TPB, 0, stream>>>(T2, I_lng, I_lnb, XN);
    // 5) xz = w_in * xn
    wmma_gemm<<<gemm_grid(2 * DIN, NTOT), TPB, 0, stream>>>(
        I_win, CDIM, XN, XN, 1 << 30, NTOT, nullptr, nullptr, XZ, NTOT,
        2 * DIN, NTOT, CDIM, 0, 0);

    // 6) two mamba directions (sequential, shared scratch)
    float* YDIR[2] = {YF, YB};
    for (int dir = 0; dir < 2; ++dir) {
        conv_silu<<<(DIN * NTOT + TPB - 1) / TPB, TPB, 0, stream>>>(
            XIN, I_wconv[dir], I_bconv[dir], XC, dir);
        // proj = w_xproj * xc   (M=64)
        wmma_gemm<<<gemm_grid(64, NTOT), TPB, 0, stream>>>(
            I_wxproj[dir], DIN, XC, XC, 1 << 30, NTOT, nullptr, nullptr, PROJ, NTOT,
            64, NTOT, DIN, 0, 0);
        // delta = softplus(w_dt * dt + b_dt)   (K=32, dt = proj rows 0..31)
        wmma_gemm<<<gemm_grid(DIN, NTOT), TPB, 0, stream>>>(
            I_wdt[dir], DTRK, PROJ, PROJ, 1 << 30, NTOT, I_bdt[dir], nullptr, DELTA, NTOT,
            DIN, NTOT, DTRK, 3, 0);
        scan_kernel<<<(BB * DIN + TPB - 1) / TPB, TPB, 0, stream>>>(
            XC, DELTA, PROJ, I_Alog[dir], I_D[dir], YDIR[dir]);
    }
    // 7) ycomb = (yf + rev(yb)) * silu(z)
    combine_kernel<<<(DIN * NTOT + TPB - 1) / TPB, TPB, 0, stream>>>(YF, YB, Z, YCOMB);
    // 8) m_out = w_out * ycomb + b_out
    wmma_gemm<<<gemm_grid(CDIM, NTOT), TPB, 0, stream>>>(
        I_wout, DIN, YCOMB, YCOMB, 1 << 30, NTOT, I_bout, nullptr, MOUT, NTOT,
        CDIM, NTOT, DIN, 0, 0);
    // 9) low = (t2 + m_out) * mask   (in-place into T2)
    low_kernel<<<(CDIM * NTOT + TPB - 1) / TPB, TPB, 0, stream>>>(T2, MOUT, I_mask);

    // 10) high2 = relu(w_c1 * [low; high] + b_c1)
    wmma_gemm<<<gemm_grid(DIN, NTOT), TPB, 0, stream>>>(
        I_wc1, 2 * CDIM, LOW, HIGH, CDIM, NTOT, I_bc1, nullptr, HIGH2, NTOT,
        DIN, NTOT, 2 * CDIM, 2, 0);
    // 11) low2 = relu(w_c2 * [high; low] + b_c2)
    wmma_gemm<<<gemm_grid(DIN, NTOT), TPB, 0, stream>>>(
        I_wc2, 2 * CDIM, HIGH, LOW, CDIM, NTOT, I_bc2, nullptr, LOW2, NTOT,
        DIN, NTOT, 2 * CDIM, 2, 0);
    // 12) out0 = relu(w_c3 * [high2; low2] + b_c3)
    wmma_gemm<<<gemm_grid(CDIM, NTOT), TPB, 0, stream>>>(
        I_wc3, 2 * DIN, HIGH2, LOW2, DIN, NTOT, I_bc3, nullptr, OUT0, NTOT,
        CDIM, NTOT, 2 * DIN, 2, 0);

    // 13) anti-collapse per batch
    rownorm_kernel<<<BB * CDIM, TPB, 0, stream>>>(OUT0, RNINV);
    for (int b = 0; b < BB; ++b) {
        const float* Xb = OUT0 + (size_t)b * TT;
        // G = (Xb . Xb^T) * rninv[m]*rninv[n]     (TEMP = 1)
        wmma_gemm<<<gemm_grid(CDIM, CDIM), TPB, 0, stream>>>(
            Xb, NTOT, nullptr, nullptr, 0, NTOT, nullptr, RNINV + (size_t)b * CDIM,
            G, CDIM, CDIM, CDIM, TT, 0, 1);
        softmax_rows<<<CDIM, TPB, 0, stream>>>(G);
        // out1 = (1+S)*out0 - S*(sim . out0)
        wmma_gemm<<<gemm_grid(CDIM, TT), TPB, 0, stream>>>(
            G, CDIM, Xb, Xb, 1 << 30, NTOT, nullptr, Xb,
            OUT1 + (size_t)b * TT, NTOT, CDIM, TT, CDIM, 0, 2);
    }

    // 14) final channel norm -> d_out ; then mask tail
    final_norm<<<BB * CDIM, TPB, 0, stream>>>(OUT1, I_cng, I_cnb, outf);
    mask_out_kernel<<<(BB * TT + TPB - 1) / TPB, TPB, 0, stream>>>(
        I_mask, outf + (size_t)BB * CDIM * TT);
}
